// TransformerBlock_45354854645817
// MI455X (gfx1250) — matrix operand order repaired and checked by reading, not compiled, not run on hardware
//
#include <hip/hip_runtime.h>
#include <hip/hip_bf16.h>
#include <math.h>

// ---------------------------------------------------------------------------
// Transformer block forward for gfx1250 (MI455X).
//  - Large GEMMs: bf16 3-term split (A_hi*B_hi + A_hi*B_lo + A_lo*B_hi) via
//    V_WMMA_F32_16X16X32_BF16 -> fp32-class accuracy at bf16 matrix rate.
//  - Attention (small FLOPs, precision-critical): V_WMMA_F32_16X16X4_F32.
// ---------------------------------------------------------------------------

typedef __attribute__((ext_vector_type(2)))  float  v2f;
typedef __attribute__((ext_vector_type(8)))  float  v8f;
typedef __attribute__((ext_vector_type(16))) __bf16 v16bf;

#define WMMA_F32(a, b, c) \
    __builtin_amdgcn_wmma_f32_16x16x4_f32(false, (a), false, (b), (short)0, (c), false, false)
#define WMMA_BF16(a, b, c) \
    __builtin_amdgcn_wmma_f32_16x16x32_bf16(false, (a), false, (b), (short)0, (c), false, false)

// ---- bf16 helpers (RNE), branch-free ----
__device__ __forceinline__ unsigned short f2bf_rne(float f) {
    unsigned int u = __float_as_uint(f);
    u += 0x7fffu + ((u >> 16) & 1u);
    return (unsigned short)(u >> 16);
}
__device__ __forceinline__ float bf2f(unsigned short h) {
    return __uint_as_float((unsigned int)h << 16);
}

// ---------------------------------------------------------------------------
// RMSNorm: y = x * rsqrt(mean(x^2) + eps) * g      (one block per row, C=1024)
// ---------------------------------------------------------------------------
__global__ __launch_bounds__(256)
void rmsnorm_kernel(const float* __restrict__ x, const float* __restrict__ g,
                    float* __restrict__ y, int C)
{
    const int row = blockIdx.x;
    const int tid = threadIdx.x;
    const float* xr = x + (size_t)row * C;
    float* yr = y + (size_t)row * C;

    float4 xv = *(const float4*)(xr + tid * 4);
    float ss = xv.x * xv.x + xv.y * xv.y + xv.z * xv.z + xv.w * xv.w;

    #pragma unroll
    for (int off = 16; off; off >>= 1) ss += __shfl_xor(ss, off, 32);

    __shared__ float red[8];
    if ((tid & 31) == 0) red[tid >> 5] = ss;
    __syncthreads();

    float tot = 0.f;
    #pragma unroll
    for (int i = 0; i < 8; ++i) tot += red[i];

    const float inv = rsqrtf(tot / (float)C + 1e-6f);
    float4 gv = *(const float4*)(g + tid * 4);
    float4 o;
    o.x = xv.x * inv * gv.x;
    o.y = xv.y * inv * gv.y;
    o.z = xv.z * inv * gv.z;
    o.w = xv.w * inv * gv.w;
    *(float4*)(yr + tid * 4) = o;
}

// ---------------------------------------------------------------------------
// bf16x3-split WMMA GEMM:  C[M,N] = act( A[M,K] @ W[N,K]^T ) (+ res[M,N])
// Block tile 64(M) x 128(N), 256 threads = 8 waves, each wave owns 32x32.
// K staged in LDS chunks of 32 as separate bf16 hi/lo planes.
// Fragment layouts (ISA 7.12.2, wave32, 16-bit):
//   A 16x32: lane l holds row m=l&15; lhalf=l>>4 selects K {0..7,16..23} vs
//            {8..15,24..31} -> two contiguous 16B runs at k=lhalf*8, 16+lhalf*8.
//   B 32x16: lane l holds col n=l&15; same K split as A -> two contiguous
//            16B runs at k=lhalf*8, 16+lhalf*8.
// act: 0 = identity, 1 = exact erf-GELU
// ---------------------------------------------------------------------------
#define GB_M 64
#define GB_N 128
#define GB_K 32
#define LDB  40   // padded LDS row stride in bf16 elements (80 bytes)

__device__ __forceinline__ v16bf load_afrag(const unsigned short* base, int lhalf) {
    union { v16bf v; uint4 u[2]; } t;
    t.u[0] = *(const uint4*)(base + lhalf * 8);
    t.u[1] = *(const uint4*)(base + lhalf * 8 + 16);
    return t.v;
}
__device__ __forceinline__ v16bf load_bfrag(const unsigned short* base, int lhalf) {
    union { v16bf v; uint4 u[2]; } t;
    t.u[0] = *(const uint4*)(base + lhalf * 8);
    t.u[1] = *(const uint4*)(base + lhalf * 8 + 16);
    return t.v;
}

// convert 8 consecutive floats -> hi/lo bf16 planes (8B-aligned ushort4 stores)
__device__ __forceinline__ void stage8(const float* __restrict__ src,
                                       unsigned short* dhi, unsigned short* dlo) {
    float4 v0 = *(const float4*)(src);
    float4 v1 = *(const float4*)(src + 4);
    ushort4 h0, h1, l0, l1;
    h0.x = f2bf_rne(v0.x); l0.x = f2bf_rne(v0.x - bf2f(h0.x));
    h0.y = f2bf_rne(v0.y); l0.y = f2bf_rne(v0.y - bf2f(h0.y));
    h0.z = f2bf_rne(v0.z); l0.z = f2bf_rne(v0.z - bf2f(h0.z));
    h0.w = f2bf_rne(v0.w); l0.w = f2bf_rne(v0.w - bf2f(h0.w));
    h1.x = f2bf_rne(v1.x); l1.x = f2bf_rne(v1.x - bf2f(h1.x));
    h1.y = f2bf_rne(v1.y); l1.y = f2bf_rne(v1.y - bf2f(h1.y));
    h1.z = f2bf_rne(v1.z); l1.z = f2bf_rne(v1.z - bf2f(h1.z));
    h1.w = f2bf_rne(v1.w); l1.w = f2bf_rne(v1.w - bf2f(h1.w));
    *(ushort4*)(dhi)     = h0;
    *(ushort4*)(dhi + 4) = h1;
    *(ushort4*)(dlo)     = l0;
    *(ushort4*)(dlo + 4) = l1;
}

__global__ __launch_bounds__(256)
void gemm_bf16x3(const float* __restrict__ A, const float* __restrict__ W,
                 const float* __restrict__ res, float* __restrict__ C,
                 int M, int N, int K, int act)
{
    __shared__ __align__(16) unsigned short As_hi[GB_M][LDB];
    __shared__ __align__(16) unsigned short As_lo[GB_M][LDB];
    __shared__ __align__(16) unsigned short Ws_hi[GB_N][LDB];
    __shared__ __align__(16) unsigned short Ws_lo[GB_N][LDB];

    const int tid   = threadIdx.x;
    const int lane  = tid & 31;
    const int wave  = tid >> 5;     // 0..7
    const int wm    = wave >> 2;    // 0..1 : 32-row slab
    const int wn    = wave & 3;     // 0..3 : 32-col slab
    const int lhalf = lane >> 4;    // 0/1
    const int lmod  = lane & 15;

    const int m0 = blockIdx.y * GB_M;
    const int n0 = blockIdx.x * GB_N;

    v8f acc[2][2];
    #pragma unroll
    for (int i = 0; i < 2; ++i)
        #pragma unroll
        for (int j = 0; j < 2; ++j)
            acc[i][j] = (v8f){0.f, 0.f, 0.f, 0.f, 0.f, 0.f, 0.f, 0.f};

    const int lr = tid >> 2;          // 0..63
    const int lc = (tid & 3) * 8;     // 0,8,16,24

    for (int k0 = 0; k0 < K; k0 += GB_K) {
        // ---- stage + split A (64x32) and W (128x32) tiles ----
        stage8(A + (size_t)(m0 + lr) * K + k0 + lc, &As_hi[lr][lc], &As_lo[lr][lc]);
        stage8(W + (size_t)(n0 + lr) * K + k0 + lc, &Ws_hi[lr][lc], &Ws_lo[lr][lc]);
        stage8(W + (size_t)(n0 + lr + 64) * K + k0 + lc,
               &Ws_hi[lr + 64][lc], &Ws_lo[lr + 64][lc]);
        __syncthreads();

        const v16bf ah0 = load_afrag(&As_hi[wm * 32 + lmod][0], lhalf);
        const v16bf al0 = load_afrag(&As_lo[wm * 32 + lmod][0], lhalf);
        const v16bf ah1 = load_afrag(&As_hi[wm * 32 + 16 + lmod][0], lhalf);
        const v16bf al1 = load_afrag(&As_lo[wm * 32 + 16 + lmod][0], lhalf);
        const v16bf bh0 = load_bfrag(&Ws_hi[wn * 32 + lmod][0], lhalf);
        const v16bf bl0 = load_bfrag(&Ws_lo[wn * 32 + lmod][0], lhalf);
        const v16bf bh1 = load_bfrag(&Ws_hi[wn * 32 + 16 + lmod][0], lhalf);
        const v16bf bl1 = load_bfrag(&Ws_lo[wn * 32 + 16 + lmod][0], lhalf);

        acc[0][0] = WMMA_BF16(al0, bh0, acc[0][0]);
        acc[0][0] = WMMA_BF16(ah0, bl0, acc[0][0]);
        acc[0][0] = WMMA_BF16(ah0, bh0, acc[0][0]);

        acc[0][1] = WMMA_BF16(al0, bh1, acc[0][1]);
        acc[0][1] = WMMA_BF16(ah0, bl1, acc[0][1]);
        acc[0][1] = WMMA_BF16(ah0, bh1, acc[0][1]);

        acc[1][0] = WMMA_BF16(al1, bh0, acc[1][0]);
        acc[1][0] = WMMA_BF16(ah1, bl0, acc[1][0]);
        acc[1][0] = WMMA_BF16(ah1, bh0, acc[1][0]);

        acc[1][1] = WMMA_BF16(al1, bh1, acc[1][1]);
        acc[1][1] = WMMA_BF16(ah1, bl1, acc[1][1]);
        acc[1][1] = WMMA_BF16(ah1, bh1, acc[1][1]);
        __syncthreads();
    }

    // ---- epilogue: D layout — vgpr v: m = v + 8*lhalf, n = lmod ----
    #pragma unroll
    for (int mi = 0; mi < 2; ++mi) {
        #pragma unroll
        for (int ni = 0; ni < 2; ++ni) {
            #pragma unroll
            for (int v = 0; v < 8; ++v) {
                const int gm = m0 + wm * 32 + mi * 16 + v + 8 * lhalf;
                const int gn = n0 + wn * 32 + ni * 16 + lmod;
                float val = acc[mi][ni][v];
                if (act == 1) {
                    val = 0.5f * val * (1.0f + erff(val * 0.7071067811865476f));
                }
                if (res) val += res[(size_t)gm * N + gn];
                C[(size_t)gm * N + gn] = val;
            }
        }
    }
}

// ---------------------------------------------------------------------------
// Fused causal attention for one (b,h) and 16 query rows per block.
//   scores = (Q K^T) * scale  -> causal mask -> softmax -> * head_mask
//   writes the full attn row slab (zero tail included) to attn_out
//   ctx = attn @ V  (split-K over waves, LDS reduction)
// qkv layout: [B,T,3,H,D] row = 3*H*D = 3072 floats.
// ---------------------------------------------------------------------------
__global__ __launch_bounds__(256)
void attn_fused_kernel(const float* __restrict__ qkv,
                       const float* __restrict__ head_mask,
                       float* __restrict__ attn_out,
                       float* __restrict__ ctx,
                       int B, int T, int H, int Dh)
{
    const int bh = blockIdx.x;
    const int b  = bh / H;
    const int h  = bh % H;
    const int t0 = blockIdx.y * 16;
    const int ncols  = t0 + 16;        // causal: need columns [0, t0+16)
    const int ntiles = ncols >> 4;

    __shared__ float Qs[16][68];          // 16 x 64 Q tile, padded
    __shared__ float Ss[16][1028];        // 16 x T score/prob buffer, padded
    __shared__ float Ored[8][16][16];     // per-wave partial outputs

    const int tid   = threadIdx.x;
    const int lane  = tid & 31;
    const int wave  = tid >> 5;
    const int lhalf = lane >> 4;
    const int lmod  = lane & 15;

    const size_t row3 = (size_t)3 * H * Dh;  // 3072
    const float* qbase = qkv + (size_t)(b * T + t0) * row3 + h * Dh;

    // ---- load Q tile (16 x 64) ----
    {
        const int r = tid >> 4;            // 0..15
        const int c = (tid & 15) * 4;      // 0..60
        float4 qv = *(const float4*)(qbase + (size_t)r * row3 + c);
        Qs[r][c + 0] = qv.x; Qs[r][c + 1] = qv.y;
        Qs[r][c + 2] = qv.z; Qs[r][c + 3] = qv.w;
    }
    __syncthreads();

    const float scale = rsqrtf((float)Dh);

    // ---- phase 1: scores = Q K^T, masked, into LDS ----
    for (int nt = wave; nt < ntiles; nt += 8) {
        const int s0 = nt * 16;
        v8f acc = (v8f){0.f, 0.f, 0.f, 0.f, 0.f, 0.f, 0.f, 0.f};
        const float* kbase = qkv + (size_t)(b * T + s0 + lmod) * row3 + H * Dh + h * Dh;
        #pragma unroll
        for (int kk = 0; kk < 16; ++kk) {
            const int ka = kk * 4 + 2 * lhalf;
            v2f a, bb;
            a.x  = Qs[lmod][ka];   a.y  = Qs[lmod][ka + 1];
            bb.x = kbase[ka];      bb.y = kbase[ka + 1];
            acc = WMMA_F32(a, bb, acc);
        }
        #pragma unroll
        for (int v = 0; v < 8; ++v) {
            const int m = v + 8 * lhalf;
            const int t = t0 + m;
            const int s = s0 + lmod;
            Ss[m][s] = (s > t) ? -__builtin_inff() : acc[v] * scale;
        }
    }
    __syncthreads();

    // ---- phase 2: row softmax + head_mask, write attn slab ----
    const float hm = head_mask[h];
    for (int r = wave; r < 16; r += 8) {
        float mx = -__builtin_inff();
        for (int s = lane; s < ncols; s += 32) mx = fmaxf(mx, Ss[r][s]);
        #pragma unroll
        for (int off = 16; off; off >>= 1) mx = fmaxf(mx, __shfl_xor(mx, off, 32));

        float sum = 0.f;
        for (int s = lane; s < ncols; s += 32) {
            float e = expf(Ss[r][s] - mx);
            Ss[r][s] = e;
            sum += e;
        }
        #pragma unroll
        for (int off = 16; off; off >>= 1) sum += __shfl_xor(sum, off, 32);

        const float inv = hm / sum;
        const int t = t0 + r;
        float* arow = attn_out + ((size_t)bh * T + t) * (size_t)T;
        for (int s = lane; s < ncols; s += 32) {
            float p = Ss[r][s] * inv;
            Ss[r][s] = p;          // keep (already * head_mask) for attn @ V
            arow[s] = p;
        }
        for (int s = ncols + lane; s < T; s += 32) arow[s] = 0.f;  // zero tail
    }
    __syncthreads();

    // ---- phase 3: ctx = P @ V.  4 d-tiles; waves split the K(S) dimension ----
    const int d0 = (wave & 3) * 16;
    v8f oacc = (v8f){0.f, 0.f, 0.f, 0.f, 0.f, 0.f, 0.f, 0.f};
    for (int st = (wave >> 2); st < ntiles; st += 2) {
        const int s0 = st * 16;
        #pragma unroll
        for (int kk = 0; kk < 4; ++kk) {
            const int ka = kk * 4 + 2 * lhalf;
            v2f a, bb;
            a.x = Ss[lmod][s0 + ka];
            a.y = Ss[lmod][s0 + ka + 1];
            const float* v0 = qkv + (size_t)(b * T + s0 + ka) * row3
                              + 2 * H * Dh + h * Dh + d0 + lmod;
            bb.x = v0[0];
            bb.y = v0[row3];       // next V row (k+1)
            oacc = WMMA_F32(a, bb, oacc);
        }
    }
    #pragma unroll
    for (int v = 0; v < 8; ++v) Ored[wave][v + 8 * lhalf][lmod] = oacc[v];
    __syncthreads();

    if (wave < 4) {
        #pragma unroll
        for (int v = 0; v < 8; ++v) {
            const int m = v + 8 * lhalf;
            float val = Ored[wave][m][lmod] + Ored[wave + 4][m][lmod];
            ctx[(size_t)(b * T + t0 + m) * (H * Dh) + h * Dh + wave * 16 + lmod] = val;
        }
    }
}

// ---------------------------------------------------------------------------
// Launcher
// inputs: x, g1, g2, w_qkv, w_out, w_fc1, w_fc2, head_mask
// d_out : [ x_final (4M floats) | attn (64M floats) ]
// ---------------------------------------------------------------------------
extern "C" void kernel_launch(void* const* d_in, const int* in_sizes, int n_in,
                              void* d_out, int out_size, void* d_ws, size_t ws_size,
                              hipStream_t stream)
{
    (void)in_sizes; (void)n_in; (void)out_size; (void)ws_size;

    const int B = 4, T = 1024, C = 1024, H = 16, Dh = 64;
    const int M = B * T;          // 4096 token rows

    const float* x    = (const float*)d_in[0];
    const float* g1   = (const float*)d_in[1];
    const float* g2   = (const float*)d_in[2];
    const float* wqkv = (const float*)d_in[3];
    const float* wout = (const float*)d_in[4];
    const float* wfc1 = (const float*)d_in[5];
    const float* wfc2 = (const float*)d_in[6];
    const float* hmsk = (const float*)d_in[7];

    float* out_x  = (float*)d_out;
    float* attn   = out_x + (size_t)M * C;          // 4,194,304 offset

    // workspace layout (floats), with reuse of dead buffers:
    float* ws   = (float*)d_ws;
    float* h    = ws;                               // 4M   [dead after qkv gemm]
    float* qkv  = ws + 4194304;                     // 12M  [dead after attention]
    float* ctx  = ws + 16777216;                    // 4M   [dead after out proj]
    float* x2   = ws + 20971520;                    // 4M
    float* h2   = ws + 16777216;                    // reuses ctx space
    float* a1   = ws;                               // 16M, reuses h+qkv space

    // 1) h = rmsnorm(x, g1)
    rmsnorm_kernel<<<M, 256, 0, stream>>>(x, g1, h, C);

    // 2) qkv = h @ w_qkv^T                     [4096 x 3072]
    gemm_bf16x3<<<dim3(3 * C / GB_N, M / GB_M), 256, 0, stream>>>(
        h, wqkv, nullptr, qkv, M, 3 * C, C, 0);

    // 3) fused attention -> attn slab + ctx    [4096 x 1024]
    attn_fused_kernel<<<dim3(B * H, T / 16), 256, 0, stream>>>(
        qkv, hmsk, attn, ctx, B, T, H, Dh);

    // 4) x2 = ctx @ w_out^T + x                [4096 x 1024]
    gemm_bf16x3<<<dim3(C / GB_N, M / GB_M), 256, 0, stream>>>(
        ctx, wout, x, x2, M, C, C, 0);

    // 5) h2 = rmsnorm(x2, g2)
    rmsnorm_kernel<<<M, 256, 0, stream>>>(x2, g2, h2, C);

    // 6) a1 = gelu(h2 @ w_fc1^T)               [4096 x 4096]
    gemm_bf16x3<<<dim3(4 * C / GB_N, M / GB_M), 256, 0, stream>>>(
        h2, wfc1, nullptr, a1, M, 4 * C, C, 1);

    // 7) out_x = a1 @ w_fc2^T + x2             [4096 x 1024]
    gemm_bf16x3<<<dim3(C / GB_N, M / GB_M), 256, 0, stream>>>(
        a1, wfc2, x2, out_x, M, C, 4 * C, 0);
}
